// LSTMNet_45243185496338
// MI455X (gfx1250) — matrix operand order repaired and checked by reading, not compiled, not run on hardware
//
#include <hip/hip_runtime.h>
#include <hip/hip_bf16.h>

// ---------------------------------------------------------------------------
// LSTMNet: out = ((h0_final @ Wo.T + bo) @ Wc.T + bc)
// Only LSTM layer 0 is live (layers 1/2 are dead code in the reference).
// One persistent workgroup per 16-row batch tile; weights resident in LDS
// as bf16; gates via v_wmma_f32_16x16x32_bf16 with dual accumulator chains;
// x_{t+1} register-prefetched behind the gate compute; 2 barriers/step.
// ---------------------------------------------------------------------------

typedef __attribute__((ext_vector_type(16))) __bf16 v16bf;
typedef __attribute__((ext_vector_type(8)))  __bf16 v8bf;
typedef __attribute__((ext_vector_type(8)))  float  v8f;

#define B_TOT  256
#define T_LEN  512
#define IN_D   128
#define HID    100
#define G4     400    // 4*HID
#define KPAD   128    // HID padded to WMMA K granularity (4 x 32)
#define TILE_B 16
#define NTILES 25     // 400 / 16
#define EMB_D  64
#define NCLS   10

union ABfrag { v16bf v; struct { v8bf lo, hi; } p; };

__device__ __forceinline__ float fast_sig(float x) {
  // v_exp_f32 + v_rcp_f32
  return __builtin_amdgcn_rcpf(1.0f + __expf(-x));
}
__device__ __forceinline__ float fast_tanh(float x) {
  // tanh(x) = 2*sigmoid(2x) - 1
  return 2.0f * __builtin_amdgcn_rcpf(1.0f + __expf(-2.0f * x)) - 1.0f;
}

__global__ __launch_bounds__(256, 1)
void lstm0_head_kernel(const float* __restrict__ x,
                       const float* __restrict__ Wih,
                       const float* __restrict__ Whh,
                       const float* __restrict__ bih,
                       const float* __restrict__ bhh,
                       const float* __restrict__ Wo,
                       const float* __restrict__ bo,
                       const float* __restrict__ Wc,
                       const float* __restrict__ bc,
                       float* __restrict__ out)
{
  // ---- LDS residency (~245 KB of the 320 KB WGP pool) ----
  __shared__ __align__(32) __bf16 sWih[G4][KPAD];   // 102.4 KB
  __shared__ __align__(32) __bf16 sWhh[G4][KPAD];   // 102.4 KB (cols >=100 zero)
  __shared__ float  sBias[G4];                      //   1.6 KB (bih+bhh)
  __shared__ __align__(32) __bf16 sX[TILE_B][KPAD]; //   4.0 KB  x_t tile
  __shared__ __align__(32) __bf16 sH[TILE_B][KPAD]; //   4.0 KB  h state (pad=0)
  __shared__ float  sC[TILE_B][HID];                //   6.4 KB  c state
  __shared__ float  sG[TILE_B][G4];                 //  25.6 KB  gate staging
  __shared__ float  sE[TILE_B][EMB_D];              //   4.0 KB  emb staging

  const int tid  = threadIdx.x;
  const int lane = tid & 31;
  const int wv   = __builtin_amdgcn_readfirstlane(tid >> 5);  // scalar wave id
  const int b0   = blockIdx.x * TILE_B;

  // ---- stage weights: f32 global -> bf16 LDS (zero-pad Whh K to 128) ----
  for (int idx = tid; idx < G4 * KPAD; idx += 256) {
    const int r = idx >> 7, c = idx & 127;
    sWih[r][c] = (__bf16)Wih[idx];                       // Wih is [400][128]
    sWhh[r][c] = (c < HID) ? (__bf16)Whh[r * HID + c] : (__bf16)0.0f;
  }
  for (int idx = tid; idx < G4; idx += 256) sBias[idx] = bih[idx] + bhh[idx];
  for (int idx = tid; idx < TILE_B * KPAD; idx += 256)
    sH[idx >> 7][idx & 127] = (__bf16)0.0f;              // h0 = 0 (incl. pad)
  for (int idx = tid; idx < TILE_B * HID; idx += 256)
    sC[idx / HID][idx % HID] = 0.0f;                     // c0 = 0

  // Lane mapping used by this kernel (16-bit operands, wave32; A/B share a contiguous K order):
  //  A (16xK): row = lane&15, 16 contiguous K at (lane>>4)*16
  //  B (Kx16): col = lane&15, 16 contiguous K at (lane>>4)*16
  //  C/D f32 : row = (lane>>4)*8 + vgpr, col = lane&15
  const int ar  = lane & 15;
  const int akb = (lane >> 4) * 16;
  const int bn  = lane & 15;
  const int bkh = (lane >> 4) * 16;

  const int xr = tid >> 4;           // staging row 0..15
  const int xc = (tid & 15) * 8;     // staging col 0..120 step 8
  const float* pxrow = x + ((size_t)(b0 + xr) * T_LEN) * IN_D + xc;

  // ---- stage x_0 ----
  {
    float4 a = *(const float4*)(pxrow);
    float4 b = *(const float4*)(pxrow + 4);
    v8bf xb;
    xb[0] = (__bf16)a.x; xb[1] = (__bf16)a.y; xb[2] = (__bf16)a.z; xb[3] = (__bf16)a.w;
    xb[4] = (__bf16)b.x; xb[5] = (__bf16)b.y; xb[6] = (__bf16)b.z; xb[7] = (__bf16)b.w;
    *(v8bf*)&sX[xr][xc] = xb;
  }
  __syncthreads();

  for (int t = 0; t < T_LEN; ++t) {
    // ---- issue x_{t+1} global loads now; consumed after the gate barrier ----
    float4 xn0, xn1;
    if (t + 1 < T_LEN) {
      const float* px = pxrow + (size_t)(t + 1) * IN_D;
      xn0 = *(const float4*)px;
      xn1 = *(const float4*)(px + 4);
      if (t + 2 < T_LEN)
        __builtin_prefetch(px + IN_D, 0, 0);   // warm t+2 (global_prefetch_b8)
    }

    // ---- A fragments: x_t and h_{t-1}, reused across all N-tiles ----
    ABfrag ax[4], ah[4];
    #pragma unroll
    for (int k = 0; k < 4; ++k) {
      const __bf16* p = &sX[ar][k * 32 + akb];
      ax[k].p.lo = *(const v8bf*)p;
      ax[k].p.hi = *(const v8bf*)(p + 8);
      const __bf16* q = &sH[ar][k * 32 + akb];
      ah[k].p.lo = *(const v8bf*)q;
      ah[k].p.hi = *(const v8bf*)(q + 8);
    }

    // ---- gates[16,400] = x_t @ Wih^T + h @ Whh^T + bias ----
    // Two independent accumulator chains interleaved to hide WMMA latency.
    for (int nt = wv; nt < NTILES; nt += 8) {   // scalar loop: EXEC stays full
      const float bv = sBias[nt * 16 + bn];
      v8f acc0 = {bv, bv, bv, bv, bv, bv, bv, bv};
      v8f acc1 = {0.f, 0.f, 0.f, 0.f, 0.f, 0.f, 0.f, 0.f};
      #pragma unroll
      for (int k = 0; k < 4; ++k) {
        v16bf bw0 = *(const v16bf*)&sWih[nt * 16 + bn][k * 32 + bkh];
        acc0 = __builtin_amdgcn_wmma_f32_16x16x32_bf16(
            false, ax[k].v, false, bw0, (short)0, acc0, false, false);
        v16bf bw1 = *(const v16bf*)&sWhh[nt * 16 + bn][k * 32 + bkh];
        acc1 = __builtin_amdgcn_wmma_f32_16x16x32_bf16(
            false, ah[k].v, false, bw1, (short)0, acc1, false, false);
      }
      v8f acc = acc0 + acc1;
      #pragma unroll
      for (int v = 0; v < 8; ++v)
        sG[(lane >> 4) * 8 + v][nt * 16 + bn] = acc[v];
    }
    __syncthreads();   // gates visible; sX/sH reads complete

    // ---- pointwise gate math (i,f,g,o) in f32; h requantized to bf16 ----
    for (int idx = tid; idx < TILE_B * HID; idx += 256) {
      const int b = idx / HID, j = idx - b * HID;
      const float gi = sG[b][j];
      const float gf = sG[b][HID + j];
      const float gg = sG[b][2 * HID + j];
      const float go = sG[b][3 * HID + j];
      const float c  = fast_sig(gf) * sC[b][j] + fast_sig(gi) * fast_tanh(gg);
      const float h  = fast_sig(go) * fast_tanh(c);
      sC[b][j] = c;
      sH[b][j] = (__bf16)h;   // pad cols 100..127 never touched -> stay 0
    }

    // ---- commit prefetched x_{t+1} tile (safe: sX reads done pre-barrier) ----
    if (t + 1 < T_LEN) {
      v8bf xb;
      xb[0] = (__bf16)xn0.x; xb[1] = (__bf16)xn0.y;
      xb[2] = (__bf16)xn0.z; xb[3] = (__bf16)xn0.w;
      xb[4] = (__bf16)xn1.x; xb[5] = (__bf16)xn1.y;
      xb[6] = (__bf16)xn1.z; xb[7] = (__bf16)xn1.w;
      *(v8bf*)&sX[xr][xc] = xb;
    }
    __syncthreads();   // h, c, x_{t+1} ready; sG free for next step
  }

  // ---- head: emb = hT @ Wo.T + bo ; logit = emb @ Wc.T + bc (tiny) ----
  for (int idx = tid; idx < TILE_B * EMB_D; idx += 256) {
    const int b = idx >> 6, e = idx & 63;
    float s = bo[e];
    for (int j = 0; j < HID; ++j) s += (float)sH[b][j] * Wo[e * HID + j];
    sE[b][e] = s;
  }
  __syncthreads();
  for (int idx = tid; idx < TILE_B * NCLS; idx += 256) {
    const int b = idx / NCLS, c = idx - b * NCLS;
    float s = bc[c];
    #pragma unroll
    for (int e = 0; e < EMB_D; ++e) s += sE[b][e] * Wc[c * EMB_D + e];
    out[(size_t)(b0 + b) * NCLS + c] = s;
  }
}

extern "C" void kernel_launch(void* const* d_in, const int* in_sizes, int n_in,
                              void* d_out, int out_size, void* d_ws, size_t ws_size,
                              hipStream_t stream) {
  // setup_inputs order: x, Wih0, Whh0, bih0, bhh0, (layers 1-2 unused),
  //                     Wo(13), bo(14), Wc(15), bc(16)
  const float* x    = (const float*)d_in[0];
  const float* Wih0 = (const float*)d_in[1];
  const float* Whh0 = (const float*)d_in[2];
  const float* bih0 = (const float*)d_in[3];
  const float* bhh0 = (const float*)d_in[4];
  const float* Wo   = (const float*)d_in[13];
  const float* bo   = (const float*)d_in[14];
  const float* Wc   = (const float*)d_in[15];
  const float* bc   = (const float*)d_in[16];
  float* out = (float*)d_out;

  lstm0_head_kernel<<<B_TOT / TILE_B, 256, 0, stream>>>(
      x, Wih0, Whh0, bih0, bhh0, Wo, bo, Wc, bc, out);
}
